// CapsNet_22376779612401
// MI455X (gfx1250) — matrix operand order repaired and checked by reading, not compiled, not run on hardware
//
#include <hip/hip_runtime.h>
#include <hip/hip_bf16.h>
#include <math.h>

typedef __attribute__((ext_vector_type(16))) _Float16 v16h;
typedef __attribute__((ext_vector_type(8)))  _Float16 v8h;
typedef __attribute__((ext_vector_type(8)))  float    v8f;
typedef __attribute__((ext_vector_type(4)))  float    v4f;

#define BATCH   256
#define ROUTES  1152
#define CAPS    10
#define ODIM    16
#define IDIM    8
#define PCK     20736          // 256*81
#define KC      64             // k-chunk staged in LDS for pcconv

#if defined(__gfx1250__) && __has_builtin(__builtin_amdgcn_global_load_async_to_lds_b32)
#define HAVE_ASYNC_LDS 1
#else
#define HAVE_ASYNC_LDS 0
#endif

#if HAVE_ASYNC_LDS
typedef __attribute__((address_space(1))) int* gint_p;
typedef __attribute__((address_space(3))) int* lint_p;
__device__ __forceinline__ void async_g2l_b32(const _Float16* g, _Float16* l) {
  __builtin_amdgcn_global_load_async_to_lds_b32((gint_p)(_Float16*)g, (lint_p)l, 0, 0);
}
template <int N>
__device__ __forceinline__ void wait_asyncN() {
#if __has_builtin(__builtin_amdgcn_s_wait_asynccnt)
  __builtin_amdgcn_s_wait_asynccnt(N);
#else
  if constexpr (N == 0) asm volatile("s_wait_asynccnt 0x0" ::: "memory");
  else                  asm volatile("s_wait_asynccnt 0x2" ::: "memory");
#endif
}
#endif

// ---------------------------------------------------------------------------
// CDNA5 WMMA f16 fragment layouts (ISA 7.12.2, wave32), h = lane>>4:
//  A 16x32: lane row M=lane%16; halves = [k0+h*16+0..15]  (contiguous k order chosen here, same as B; not the ISA table order)
//  B 32x16: lane col N=lane%16; halves = [k0+h*16+0..15]  (contiguous k order chosen here, same as A; not the ISA table order)
//  C 16x16 f32: lane col N=lane%16; VGPR r -> row r+8h
// ---------------------------------------------------------------------------
__device__ __forceinline__ v16h load_a_frag_f16(const _Float16* row, int k0, int h) {
  const _Float16* p = row + k0 + h * 16;
  v8h lo = *(const v8h*)(p);
  v8h hi = *(const v8h*)(p + 8);
  v16h f;
#pragma unroll
  for (int i = 0; i < 8; ++i) { f[i] = lo[i]; f[i + 8] = hi[i]; }
  return f;
}
__device__ __forceinline__ v16h load_b_frag_f16(const _Float16* row, int k0, int h) {
  const _Float16* p = row + k0 + h * 16;
  v8h lo = *(const v8h*)(p);
  v8h hi = *(const v8h*)(p + 8);
  v16h f;
#pragma unroll
  for (int i = 0; i < 8; ++i) { f[i] = lo[i]; f[i + 8] = hi[i]; }
  return f;
}
__device__ __forceinline__ v16h load_a_frag_f32(const float* row, int k0, int h) {
  const float* p = row + k0 + h * 16;
  v4f a0 = *(const v4f*)(p);
  v4f a1 = *(const v4f*)(p + 4);
  v4f a2 = *(const v4f*)(p + 8);
  v4f a3 = *(const v4f*)(p + 12);
  v16h f;
#pragma unroll
  for (int i = 0; i < 4; ++i) {
    f[i]      = (_Float16)a0[i];
    f[i + 4]  = (_Float16)a1[i];
    f[i + 8]  = (_Float16)a2[i];
    f[i + 12] = (_Float16)a3[i];
  }
  return f;
}

// compile-time conv1 patch offset: kb -> ky*28+kx (0 if pad)
__device__ constexpr int poff_c(int kb) {
  int ky = kb / 9, kx = kb - ky * 9;
  return (kb < 81) ? (ky * 28 + kx) : 0;
}

// ---------------------------------------------------------------------------
// small utility kernels
// ---------------------------------------------------------------------------
__global__ void zerof_kernel(float* p, int n) {
  int i = blockIdx.x * blockDim.x + threadIdx.x;
  if (i < n) p[i] = 0.0f;
}

__global__ void cvt_c1w_kernel(const float* __restrict__ w, _Float16* __restrict__ wh) {
  int t = blockIdx.x * blockDim.x + threadIdx.x;   // 256*96
  if (t >= 256 * 96) return;
  int m = t / 96, k = t % 96;
  wh[t] = (k < 81) ? (_Float16)w[m * 81 + k] : (_Float16)0.0f;
}

__global__ void cvt_f32_f16_kernel(const float* __restrict__ w, _Float16* __restrict__ wh,
                                   int n) {
  for (size_t i = blockIdx.x * (size_t)blockDim.x + threadIdx.x; i < (size_t)n;
       i += (size_t)gridDim.x * blockDim.x)
    wh[i] = (_Float16)w[i];
}

// rowtab[k] = ci*400 + ky*20 + kx   (block-independent part of patch row addr)
__global__ void build_rowtab_kernel(int* __restrict__ tab) {
  int k = blockIdx.x * blockDim.x + threadIdx.x;
  if (k >= PCK) return;
  int ci = k / 81, r2 = k - ci * 81, ky = r2 / 9, kx = r2 - ky * 9;
  tab[k] = ci * 400 + ky * 20 + kx;
}

__global__ void sumsq_kernel(const float* __restrict__ x, int n, float* __restrict__ acc) {
  __shared__ float red[256];
  float s = 0.0f;
  for (int i = blockIdx.x * blockDim.x + threadIdx.x; i < n; i += gridDim.x * blockDim.x) {
    float v = x[i]; s += v * v;
  }
  red[threadIdx.x] = s; __syncthreads();
  for (int off = 128; off > 0; off >>= 1) {
    if (threadIdx.x < off) red[threadIdx.x] += red[threadIdx.x + off];
    __syncthreads();
  }
  if (threadIdx.x == 0) atomicAdd(acc, red[0]);
}

// ---------------------------------------------------------------------------
// conv1 (9x9 s1, 1->256) implicit GEMM: M=256(oc), N=B*400, K=81->96.
// Output layout H2[oc][iy][ix][b]  (batch innermost) f16, relu applied.
// ---------------------------------------------------------------------------
__global__ void conv1_wmma_kernel(const _Float16* __restrict__ Wc,   // 256 x 96 f16
                                  const float* __restrict__ x,       // (B,1,28,28)
                                  const float* __restrict__ bias,    // 256
                                  _Float16* __restrict__ H2) {       // (256,20,20,B)
  const int lane  = threadIdx.x;
  const int ntile = blockIdx.x;       // 0..6399
  const int mtile = blockIdx.y;       // 0..15
  const int lr    = lane & 15;
  const int h     = (lane >> 4) & 1;
  const int mrow  = mtile * 16 + lr;
  const int n     = ntile * 16 + lr;
  const int b     = n / 400, rem = n % 400, oy = rem / 20, ox = rem % 20;
  const int pbase = oy * 28 + ox;
  const float* xb = x + (size_t)b * 784;
  const _Float16* arow = Wc + (size_t)mrow * 96;

  v8f c = {};
#pragma unroll
  for (int k0 = 0; k0 < 96; k0 += 32) {
    v16h a = load_a_frag_f16(arow, k0, h);
    v16h bb;
#pragma unroll
    for (int e = 0; e < 16; ++e) {
      const int plo = poff_c(k0 + e);            // compile-time constants
      const int phi = poff_c(k0 + 16 + e);
      const bool vlo = (k0 + e) < 81;
      const bool vhi = (k0 + 16 + e) < 81;
      int po = h ? phi : plo;
      float xv = xb[pbase + po];
      bool valid = h ? vhi : vlo;
      bb[e] = valid ? (_Float16)xv : (_Float16)0.0f;
    }
    c = __builtin_amdgcn_wmma_f32_16x16x32_f16(false, a, false, bb, (short)0, c,
                                               false, false);
  }
  const int ocb = mtile * 16 + 8 * h;
  v4f b0 = *(const v4f*)(bias + ocb);
  v4f b1 = *(const v4f*)(bias + ocb + 4);
  _Float16* hp = H2 + ((size_t)(ocb * 400 + rem)) * BATCH + b;   // base + imm offsets
#pragma unroll
  for (int r = 0; r < 8; ++r) {
    float bi = (r < 4) ? b0[r] : b1[r - 4];
    hp[(size_t)r * 400 * BATCH] = (_Float16)fmaxf(c[r] + bi, 0.0f);
  }
}

// ---------------------------------------------------------------------------
// primary caps conv (9x9 s2, 256->256) as per-position GEMM with LDS staging:
//   out[oc, b] = sum_k Wp[oc,k] * Patch_p[k, b],  K=20736, N=256(b), M=256(oc)
// Block: 256 thr (8 waves), fixed (pos, 16-b tile); each wave does 2 M-tiles.
// HAVE_ASYNC_LDS: double-buffered global_load_async_to_lds_b32 staging in
// [k][b] layout (ASYNCcnt-tracked, s_wait_asynccnt 2 retires the old chunk
// while the new one is in flight).  Fallback: register-pipelined coalesced
// b64 loads + transposed ds_store into [b][k] layout (vector ds_load_b128
// B-fragments).  Patch rows come from rowtab (no div/mod in the hot loop).
// ---------------------------------------------------------------------------
__global__ void pcconv_wmma_kernel(const _Float16* __restrict__ Wp,  // 256 x 20736 f16
                                   const _Float16* __restrict__ H2,  // (256,20,20,B)
                                   const int* __restrict__ rowtab,   // 20736
                                   const float* __restrict__ pb,     // 256
                                   float* __restrict__ U) {          // (B,9216)
  const int tid   = threadIdx.x;
  const int lane  = tid & 31;
  const int wave  = tid >> 5;          // 0..7
  const int ntile = blockIdx.x;        // 0..15 : 16-batch tile
  const int pos   = blockIdx.y;        // 0..35
  const int oy = pos / 6, ox = pos % 6;
  const int posoff = (oy * 2) * 20 + ox * 2;
  const int bbase  = ntile * 16;
  const int lr = lane & 15;
  const int h  = (lane >> 4) & 1;

  const _Float16* arow0 = Wp + (size_t)(wave * 16 + lr) * PCK;  // mtile = wave
  const _Float16* arow1 = arow0 + (size_t)128 * PCK;            // mtile = wave+8

  v8f acc0 = {}, acc1 = {};
  const int NCHUNK = PCK / KC;         // 324

#if HAVE_ASYNC_LDS
  // ------------- async double-buffered staging, LDS layout [k][b] ----------
  __shared__ __align__(16) _Float16 patchKB[2][KC * 16];   // 2 x 2KB
  const int d0 = tid;                  // dword ids: 512 dwords per chunk
  const int d1 = tid + 256;
  auto issue = [&](int c, int buf) {
    {
      int row = rowtab[c * KC + (d0 >> 3)] + posoff;
      const _Float16* g = H2 + (size_t)row * BATCH + bbase + (d0 & 7) * 2;
      async_g2l_b32(g, &patchKB[buf][(d0 >> 3) * 16 + (d0 & 7) * 2]);
    }
    {
      int row = rowtab[c * KC + (d1 >> 3)] + posoff;
      const _Float16* g = H2 + (size_t)row * BATCH + bbase + (d1 & 7) * 2;
      async_g2l_b32(g, &patchKB[buf][(d1 >> 3) * 16 + (d1 & 7) * 2]);
    }
  };
  issue(0, 0);
  for (int c = 0; c < NCHUNK; ++c) {
    if (c + 1 < NCHUNK) {              // overlap next chunk with this compute
      issue(c + 1, (c + 1) & 1);
      wait_asyncN<2>();                // retire chunk c only
    } else {
      wait_asyncN<0>();
    }
    __syncthreads();
    const _Float16* pk = patchKB[c & 1];
    const int k0 = c * KC;
#pragma unroll
    for (int ks = 0; ks < 2; ++ks) {
      v16h bf;
#pragma unroll
      for (int e = 0; e < 16; ++e)     // [k][b] gather (ds_load_u16)
        bf[e] = pk[(ks * 32 + h * 16 + e) * 16 + lr];
      v16h a0 = load_a_frag_f16(arow0, k0 + ks * 32, h);
      v16h a1 = load_a_frag_f16(arow1, k0 + ks * 32, h);
      acc0 = __builtin_amdgcn_wmma_f32_16x16x32_f16(false, a0, false, bf, (short)0,
                                                    acc0, false, false);
      acc1 = __builtin_amdgcn_wmma_f32_16x16x32_f16(false, a1, false, bf, (short)0,
                                                    acc1, false, false);
    }
    __syncthreads();                   // buf reuse safety (chunk c+2 -> buf c&1)
  }
#else
  // ------------- register-pipelined staging, LDS layout [b][k] -------------
  __shared__ __align__(16) _Float16 patch[16 * KC];   // 2KB
  const int kk_s = tid >> 2;           // 0..63  (k row within chunk)
  const int part = tid & 3;            // 0..3   (4 b's per thread)
  const _Float16* hsrc = H2 + bbase + part * 4;

  union U2 { uint2 u; _Float16 hh[4]; };
  U2 stage;
  stage.u = *(const uint2*)(hsrc + (size_t)(rowtab[kk_s] + posoff) * BATCH);

  for (int c = 0; c < NCHUNK; ++c) {
    __syncthreads();
#pragma unroll
    for (int j = 0; j < 4; ++j)
      patch[(part * 4 + j) * KC + kk_s] = stage.hh[j];
    __syncthreads();
    if (c + 1 < NCHUNK) {              // issue next chunk's loads early
      int row = rowtab[(c + 1) * KC + kk_s] + posoff;
      stage.u = *(const uint2*)(hsrc + (size_t)row * BATCH);
    }
    const int k0 = c * KC;
#pragma unroll
    for (int ks = 0; ks < 2; ++ks) {
      v16h bf = load_b_frag_f16(&patch[lr * KC], ks * 32, h);   // 2x ds_load_b128
      v16h a0 = load_a_frag_f16(arow0, k0 + ks * 32, h);
      v16h a1 = load_a_frag_f16(arow1, k0 + ks * 32, h);
      acc0 = __builtin_amdgcn_wmma_f32_16x16x32_f16(false, a0, false, bf, (short)0,
                                                    acc0, false, false);
      acc1 = __builtin_amdgcn_wmma_f32_16x16x32_f16(false, a1, false, bf, (short)0,
                                                    acc1, false, false);
    }
  }
#endif

  const int b    = bbase + lr;
  const int ocb0 = wave * 16 + 8 * h;
  v4f p0a = *(const v4f*)(pb + ocb0);
  v4f p0b = *(const v4f*)(pb + ocb0 + 4);
  v4f p1a = *(const v4f*)(pb + ocb0 + 128);
  v4f p1b = *(const v4f*)(pb + ocb0 + 132);
  float* up0 = U + (size_t)b * 9216 + ocb0 * 36 + pos;          // base + imm offsets
  float* up1 = up0 + 128 * 36;
#pragma unroll
  for (int r = 0; r < 8; ++r) {
    float bi0 = (r < 4) ? p0a[r] : p0b[r - 4];
    float bi1 = (r < 4) ? p1a[r] : p1b[r - 4];
    up0[r * 36] = acc0[r] + bi0;
    up1[r * 36] = acc1[r] + bi1;
  }
}

// ---------------------------------------------------------------------------
// u_hat[b,r,q,o] = squash_scale * sum_i W[r,q,o,i]*U[b,r,i]   (f16 out)
// ---------------------------------------------------------------------------
__global__ void uhat_kernel(const float* __restrict__ U, const float* __restrict__ Wd,
                            const float* __restrict__ ss, _Float16* __restrict__ uh) {
  int t = blockIdx.x * blockDim.x + threadIdx.x;
  if (t >= BATCH * ROUTES * CAPS) return;
  int q = t % CAPS, br = t / CAPS, r = br % ROUTES, b = br / ROUTES;
  float s2 = ss[0];
  float scale = sqrtf(s2) / (1.0f + s2);
  const float* up = U + (size_t)b * 9216 + r * IDIM;
  v4f u0 = *(const v4f*)(up);
  v4f u1 = *(const v4f*)(up + 4);
  float u[IDIM];
#pragma unroll
  for (int i = 0; i < 4; ++i) { u[i] = u0[i] * scale; u[i + 4] = u1[i] * scale; }
  const float* w = Wd + ((size_t)(r * CAPS + q)) * (ODIM * IDIM);
  _Float16* o = uh + (((size_t)b * ROUTES + r) * CAPS + q) * ODIM;
  v8h r0, r1;
#pragma unroll
  for (int oo = 0; oo < ODIM; ++oo) {
    v4f w0 = *(const v4f*)(w + oo * IDIM);
    v4f w1 = *(const v4f*)(w + oo * IDIM + 4);
    float acc = 0.0f;
#pragma unroll
    for (int i = 0; i < 4; ++i) acc += w0[i] * u[i] + w1[i] * u[i + 4];
    if (oo < 8) r0[oo] = (_Float16)acc; else r1[oo - 8] = (_Float16)acc;
  }
  *(v8h*)(o)     = r0;
  *(v8h*)(o + 8) = r1;
}

// softmax over route axis (axis 0) of b_log (1152,10) -> c
__global__ void softmax_routes_kernel(const float* __restrict__ blog, float* __restrict__ c) {
  int q = blockIdx.x, tid = threadIdx.x;
  __shared__ float red[256];
  __shared__ float bc[2];
  float m = -1e30f;
  for (int r = tid; r < ROUTES; r += 256) m = fmaxf(m, blog[r * CAPS + q]);
  red[tid] = m; __syncthreads();
  for (int off = 128; off > 0; off >>= 1) {
    if (tid < off) red[tid] = fmaxf(red[tid], red[tid + off]);
    __syncthreads();
  }
  if (tid == 0) bc[0] = red[0];
  __syncthreads();
  m = bc[0];
  float s = 0.0f;
  for (int r = tid; r < ROUTES; r += 256) s += __expf(blog[r * CAPS + q] - m);
  red[tid] = s; __syncthreads();
  for (int off = 128; off > 0; off >>= 1) {
    if (tid < off) red[tid] += red[tid + off];
    __syncthreads();
  }
  if (tid == 0) bc[1] = red[0];
  __syncthreads();
  float inv = 1.0f / bc[1];
  for (int r = tid; r < ROUTES; r += 256)
    c[r * CAPS + q] = __expf(blog[r * CAPS + q] - m) * inv;
}

// s[b,q,o] = sum_r u_hat[b,r,q,o] * c[r,q]
__global__ void s_kernel(const _Float16* __restrict__ uh, const float* __restrict__ c,
                         float* __restrict__ s) {
  int t = blockIdx.x * blockDim.x + threadIdx.x;
  if (t >= BATCH * CAPS * ODIM) return;
  int o = t & 15, bq = t >> 4, q = bq % CAPS, b = bq / CAPS;
  const _Float16* p = uh + ((size_t)b * ROUTES * CAPS + q) * ODIM + o;
  float acc = 0.0f;
  for (int r = 0; r < ROUTES; ++r) {
    __builtin_prefetch(p + (size_t)(r + 16) * (CAPS * ODIM), 0, 1);  // global_prefetch_b8
    acc += (float)p[(size_t)r * (CAPS * ODIM)] * c[r * CAPS + q];
  }
  s[t] = acc;
}

// b_log[r,q] += mean_b sum_o u_hat[b,r,q,o] * v[b,q,o],  v = scale2 * s
__global__ void bupdate_kernel(const _Float16* __restrict__ uh, const float* __restrict__ s,
                               const float* __restrict__ ss2, float* __restrict__ blog) {
  int t = blockIdx.x * blockDim.x + threadIdx.x;
  if (t >= ROUTES * CAPS) return;
  int q = t % CAPS, r = t / CAPS;
  float s2 = ss2[0];
  float scale = sqrtf(s2) / (1.0f + s2);
  float acc = 0.0f;
  for (int b = 0; b < BATCH; ++b) {
    const _Float16* p = uh + (((size_t)b * ROUTES + r) * CAPS + q) * ODIM;
    const float* vp = s + ((size_t)b * CAPS + q) * ODIM;
    float d = 0.0f;
#pragma unroll
    for (int o = 0; o < ODIM; ++o) d += (float)p[o] * vp[o];
    acc += d;
  }
  blog[t] += acc * scale * (1.0f / (float)BATCH);
}

// final v = scale2 * s, to ws and d_out (shape (B,10,16,1) flat)
__global__ void vfinal_kernel(const float* __restrict__ s, const float* __restrict__ ss2,
                              float* __restrict__ v, float* __restrict__ dout) {
  int t = blockIdx.x * blockDim.x + threadIdx.x;
  if (t >= BATCH * CAPS * ODIM) return;
  float s2 = ss2[0];
  float scale = sqrtf(s2) / (1.0f + s2);
  float val = s[t] * scale;
  v[t] = val;
  dout[t] = val;
}

// classes[b,q] = ||v[b,q,:]||
__global__ void cls_kernel(const float* __restrict__ v, float* __restrict__ cls) {
  int t = blockIdx.x * blockDim.x + threadIdx.x;
  if (t >= BATCH * CAPS) return;
  const float* p = v + (size_t)t * ODIM;
  float s = 0.0f;
#pragma unroll
  for (int o = 0; o < ODIM; ++o) s += p[o] * p[o];
  cls[t] = sqrtf(s);
}

// per-q softmax stats over the BATCH axis (legacy softmax axis=0)
__global__ void cls_stats_kernel(const float* __restrict__ cls, float* __restrict__ qmax,
                                 float* __restrict__ qsum) {
  int q = blockIdx.x, b = threadIdx.x;   // blockDim.x == 256 == BATCH
  __shared__ float red[256];
  __shared__ float mshare;
  float x = cls[b * CAPS + q];
  red[b] = x; __syncthreads();
  for (int off = 128; off > 0; off >>= 1) {
    if (b < off) red[b] = fmaxf(red[b], red[b + off]);
    __syncthreads();
  }
  if (b == 0) { qmax[q] = red[0]; mshare = red[0]; }
  __syncthreads();
  float m = mshare;
  red[b] = __expf(x - m); __syncthreads();
  for (int off = 128; off > 0; off >>= 1) {
    if (b < off) red[b] += red[b + off];
    __syncthreads();
  }
  if (b == 0) qsum[q] = red[0];
}

// per-batch argmax over softmaxed classes; build masked flat (B,160)
__global__ void mask_kernel(const float* __restrict__ cls, const float* __restrict__ qmax,
                            const float* __restrict__ qsum, const float* __restrict__ v,
                            float* __restrict__ flat) {
  int b = blockIdx.x * blockDim.x + threadIdx.x;
  if (b >= BATCH) return;
  int best = 0; float bv = -1e30f;
#pragma unroll
  for (int q = 0; q < CAPS; ++q) {
    float p = __expf(cls[b * CAPS + q] - qmax[q]) / qsum[q];
    if (p > bv) { bv = p; best = q; }
  }
  float* fr = flat + (size_t)b * (CAPS * ODIM);
  for (int j = 0; j < CAPS * ODIM; ++j) fr[j] = 0.0f;
  const float* vp = v + ((size_t)b * CAPS + best) * ODIM;
#pragma unroll
  for (int o = 0; o < ODIM; ++o) fr[best * ODIM + o] = vp[o];
}

// ---------------------------------------------------------------------------
// decoder GEMM: out[m,n] = act(sum_k A[m,k]*Wh[n,k] + bias[n])
// A: MxK f32 (vector-loaded + cvt), Wh: NxK f16 (contiguous b128 loads).
// ---------------------------------------------------------------------------
__global__ void gemm_wmma_kernel(const float* __restrict__ A, const _Float16* __restrict__ Wh,
                                 const float* __restrict__ bias, float* __restrict__ out,
                                 int M, int N, int K, int act) {
  const int lane  = threadIdx.x;
  const int ntile = blockIdx.x;
  const int mtile = blockIdx.y;
  const int lr    = lane & 15;
  const int h     = (lane >> 4) & 1;
  const int mrow  = mtile * 16 + lr;
  const int ncol  = ntile * 16 + lr;
  const float*    arow = A  + (size_t)mrow * K;
  const _Float16* brow = Wh + (size_t)ncol * K;
  v8f c = {};
  for (int k0 = 0; k0 < K; k0 += 32) {
    v16h a = load_a_frag_f32(arow, k0, h);
    v16h b = load_b_frag_f16(brow, k0, h);
    c = __builtin_amdgcn_wmma_f32_16x16x32_f16(false, a, false, b, (short)0, c,
                                               false, false);
  }
  float bv = bias[ncol];
  float* op = out + (size_t)(mtile * 16 + 8 * h) * N + ncol;     // base + imm offsets
#pragma unroll
  for (int r = 0; r < 8; ++r) {
    float xx = c[r] + bv;
    xx = (act == 0) ? fmaxf(xx, 0.0f) : (1.0f / (1.0f + __expf(-xx)));
    op[(size_t)r * N] = xx;
  }
}

// ---------------------------------------------------------------------------
extern "C" void kernel_launch(void* const* d_in, const int* in_sizes, int n_in,
                              void* d_out, int out_size, void* d_ws, size_t ws_size,
                              hipStream_t stream) {
  (void)in_sizes; (void)n_in; (void)out_size; (void)ws_size;
  const float* x   = (const float*)d_in[0];
  const float* c1w = (const float*)d_in[1];
  const float* c1b = (const float*)d_in[2];
  const float* pcw = (const float*)d_in[3];
  const float* pcb = (const float*)d_in[4];
  const float* Wd  = (const float*)d_in[5];
  const float* d1w = (const float*)d_in[6];
  const float* d1b = (const float*)d_in[7];
  const float* d2w = (const float*)d_in[8];
  const float* d2b = (const float*)d_in[9];
  const float* d3w = (const float*)d_in[10];
  const float* d3b = (const float*)d_in[11];
  float* out_v   = (float*)d_out;
  float* out_rec = (float*)d_out + BATCH * CAPS * ODIM;

  char* wp = (char*)d_ws;
  auto alloc = [&](size_t bytes) {
    char* r = wp;
    wp += (bytes + 255) & ~(size_t)255;
    return r;
  };
  float*    ss     = (float*)alloc(2 * sizeof(float));
  float*    qmax   = (float*)alloc(CAPS * sizeof(float));
  float*    qsum   = (float*)alloc(CAPS * sizeof(float));
  float*    blog   = (float*)alloc(ROUTES * CAPS * sizeof(float));
  float*    cbuf   = (float*)alloc(ROUTES * CAPS * sizeof(float));
  float*    cls    = (float*)alloc(BATCH * CAPS * sizeof(float));
  float*    sbuf   = (float*)alloc((size_t)BATCH * CAPS * ODIM * sizeof(float));
  float*    vbuf   = (float*)alloc((size_t)BATCH * CAPS * ODIM * sizeof(float));
  float*    flat   = (float*)alloc((size_t)BATCH * CAPS * ODIM * sizeof(float));
  float*    r1     = (float*)alloc((size_t)BATCH * 512 * sizeof(float));
  float*    r2     = (float*)alloc((size_t)BATCH * 1024 * sizeof(float));
  int*      rowtab = (int*)alloc((size_t)PCK * sizeof(int));
  _Float16* Wc1h   = (_Float16*)alloc((size_t)256 * 96 * sizeof(_Float16));
  _Float16* Wpch   = (_Float16*)alloc((size_t)256 * PCK * sizeof(_Float16));
  _Float16* W1h    = (_Float16*)alloc((size_t)512 * 160 * sizeof(_Float16));
  _Float16* W2h    = (_Float16*)alloc((size_t)1024 * 512 * sizeof(_Float16));
  _Float16* W3h    = (_Float16*)alloc((size_t)784 * 1024 * sizeof(_Float16));
  _Float16* H2     = (_Float16*)alloc((size_t)256 * 400 * BATCH * sizeof(_Float16));
  float*    U      = (float*)alloc((size_t)BATCH * 9216 * sizeof(float));
  _Float16* uh     = (_Float16*)alloc((size_t)BATCH * ROUTES * CAPS * ODIM * sizeof(_Float16));

  // init + tables + weight conversions
  zerof_kernel<<<1, 32, 0, stream>>>(ss, 2);
  zerof_kernel<<<(ROUTES * CAPS + 255) / 256, 256, 0, stream>>>(blog, ROUTES * CAPS);
  build_rowtab_kernel<<<(PCK + 255) / 256, 256, 0, stream>>>(rowtab);
  cvt_c1w_kernel<<<(256 * 96 + 255) / 256, 256, 0, stream>>>(c1w, Wc1h);
  cvt_f32_f16_kernel<<<2048, 256, 0, stream>>>(pcw, Wpch, 256 * PCK);
  cvt_f32_f16_kernel<<<320, 256, 0, stream>>>(d1w, W1h, 512 * 160);
  cvt_f32_f16_kernel<<<2048, 256, 0, stream>>>(d2w, W2h, 1024 * 512);
  cvt_f32_f16_kernel<<<2048, 256, 0, stream>>>(d3w, W3h, 784 * 1024);

  // conv1 + relu -> H2 (oc, iy, ix, b) f16
  conv1_wmma_kernel<<<dim3(6400, 16), 32, 0, stream>>>(Wc1h, x, c1b, H2);

  // primary caps conv + bias -> U (b, 9216)
  pcconv_wmma_kernel<<<dim3(16, 36), 256, 0, stream>>>(Wpch, H2, rowtab, pcb, U);

  // global squash scale of u
  sumsq_kernel<<<1024, 256, 0, stream>>>(U, BATCH * 9216, &ss[0]);

  // u_hat (f16)
  uhat_kernel<<<(BATCH * ROUTES * CAPS + 255) / 256, 256, 0, stream>>>(U, Wd, &ss[0], uh);

  // dynamic routing, 3 iterations
  for (int it = 0; it < 3; ++it) {
    softmax_routes_kernel<<<CAPS, 256, 0, stream>>>(blog, cbuf);
    s_kernel<<<(BATCH * CAPS * ODIM + 255) / 256, 256, 0, stream>>>(uh, cbuf, sbuf);
    zerof_kernel<<<1, 32, 0, stream>>>(&ss[1], 1);
    sumsq_kernel<<<160, 256, 0, stream>>>(sbuf, BATCH * CAPS * ODIM, &ss[1]);
    if (it < 2) {
      bupdate_kernel<<<(ROUTES * CAPS + 255) / 256, 256, 0, stream>>>(uh, sbuf, &ss[1], blog);
    } else {
      vfinal_kernel<<<(BATCH * CAPS * ODIM + 255) / 256, 256, 0, stream>>>(sbuf, &ss[1],
                                                                           vbuf, out_v);
    }
  }

  // decoder masking (legacy batch-axis softmax + argmax)
  cls_kernel<<<(BATCH * CAPS + 255) / 256, 256, 0, stream>>>(vbuf, cls);
  cls_stats_kernel<<<CAPS, 256, 0, stream>>>(cls, qmax, qsum);
  mask_kernel<<<1, 256, 0, stream>>>(cls, qmax, qsum, vbuf, flat);

  // decoder MLP (WMMA)
  gemm_wmma_kernel<<<dim3(512 / 16, BATCH / 16), 32, 0, stream>>>(flat, W1h, d1b, r1,
                                                                  BATCH, 512, 160, 0);
  gemm_wmma_kernel<<<dim3(1024 / 16, BATCH / 16), 32, 0, stream>>>(r1, W2h, d2b, r2,
                                                                   BATCH, 1024, 512, 0);
  gemm_wmma_kernel<<<dim3(784 / 16, BATCH / 16), 32, 0, stream>>>(r2, W3h, d3b, out_rec,
                                                                  BATCH, 784, 1024, 1);
}
